// HybridNetwork_86870008528902
// MI455X (gfx1250) — matrix operand order repaired and checked by reading, not compiled, not run on hardware
//
#include <hip/hip_runtime.h>
#include <hip/hip_bf16.h>

// ---------------------------------------------------------------------------
// HybridNetwork: photonic MZI mesh (collapsed to dense complex operator) ->
// TIA clip -> PCM crossbar GEMM.  gfx1250 wave32 + v_wmma_f32_16x16x32_bf16.
// Round 2: 2x2 register-blocked WMMA GEMMs (32x32 macro-tile per wave),
// imag operand pre-negated so Re(X@C) accumulates into one WMMA chain.
// ---------------------------------------------------------------------------

#define NCH   512
#define BATCH 4096
#define T_AMP 0.99426008f          // 10^(-0.05/20)
#define GMIN  1e-6f
#define GMAX  1e-4f

typedef __attribute__((ext_vector_type(16))) __bf16 v16bf;
typedef __attribute__((ext_vector_type(8)))  float  v8f;

union FragBF { v16bf v; uint4 u[2]; };

__device__ inline unsigned short f2bf(float f) {
    union { float f; unsigned int u; } c; c.f = f;
    unsigned int u = c.u + 0x7FFFu + ((c.u >> 16) & 1u);   // round-to-nearest-even
    return (unsigned short)(u >> 16);
}

// ---------------------------------------------------------------------------
// Kernel 0: fp32 -> bf16 conversion; imag part is NEGATED so the first GEMM
// can accumulate Xr@Cr + (-Xi)@Ci into a single accumulator.
// ---------------------------------------------------------------------------
__global__ void convert_x_kernel(const float* __restrict__ xr,
                                 const float* __restrict__ xi,
                                 unsigned short* __restrict__ xr_bf,
                                 unsigned short* __restrict__ xin_bf, int n) {
    int i = blockIdx.x * blockDim.x + threadIdx.x;
    if (i < n) {
        xr_bf[i]  = f2bf(xr[i]);
        xin_bf[i] = f2bf(-xi[i]);
    }
}

// ---------------------------------------------------------------------------
// Kernel 1: G = Gmin + state*(Gmax-Gmin), stored transposed (GT[n][k]) in bf16
// so WMMA B-fragments are contiguous 128-bit loads.
// ---------------------------------------------------------------------------
__global__ void make_gt_kernel(const float* __restrict__ states,
                               unsigned short* __restrict__ gt) {
    int i = blockIdx.x * blockDim.x + threadIdx.x;      // over 512*512
    int k = i >> 9, n = i & (NCH - 1);
    gt[(size_t)n * NCH + k] = f2bf(GMIN + states[i] * (GMAX - GMIN));
}

// ---------------------------------------------------------------------------
// Kernel 2: collapse the 512-layer MZI mesh into a dense complex matrix C
// (x_out = x_in @ C).  One workgroup per basis row, row lives in LDS.
// Writes C transposed (CrT[n][k] = Re C[k][n]) in bf16 for the GEMM B path.
// ---------------------------------------------------------------------------
__global__ void mesh_build_kernel(const float* __restrict__ pe,   // [256][256]
                                  const float* __restrict__ po,   // [256][255]
                                  unsigned short* __restrict__ crt,
                                  unsigned short* __restrict__ cit) {
    __shared__ float re[NCH];
    __shared__ float im[NCH];
    const int t   = threadIdx.x;     // 0..255
    const int row = blockIdx.x;      // 0..511  (basis vector index = K index)

    re[2 * t]     = (2 * t     == row) ? 1.f : 0.f;  im[2 * t]     = 0.f;
    re[2 * t + 1] = (2 * t + 1 == row) ? 1.f : 0.f;  im[2 * t + 1] = 0.f;
    __syncthreads();

    for (int step = 0; step < 256; ++step) {
        // ---- even layer: pairs (2t, 2t+1) -------------------------------
        {
            float h = pe[step * 256 + t] * 0.5f, s, c;
            __sincosf(h, &s, &c);
            float pr = -T_AMP * s, pi = T_AMP * c;   // i*e^{i h} * T_AMP
            int i0 = 2 * t, i1 = 2 * t + 1;
            float ar = re[i0], ai = im[i0], br = re[i1], bi = im[i1];
            float tar = s * ar + c * br, tai = s * ai + c * bi;
            float tbr = c * ar - s * br, tbi = c * ai - s * bi;
            re[i0] = pr * tar - pi * tai;  im[i0] = pr * tai + pi * tar;
            re[i1] = pr * tbr - pi * tbi;  im[i1] = pr * tbi + pi * tbr;
        }
        __syncthreads();
        // ---- odd layer: pairs (2t+1, 2t+2); edges get prop loss ---------
        if (t < 255) {
            float h = po[step * 255 + t] * 0.5f, s, c;
            __sincosf(h, &s, &c);
            float pr = -T_AMP * s, pi = T_AMP * c;
            int i0 = 2 * t + 1, i1 = 2 * t + 2;
            float ar = re[i0], ai = im[i0], br = re[i1], bi = im[i1];
            float tar = s * ar + c * br, tai = s * ai + c * bi;
            float tbr = c * ar - s * br, tbi = c * ai - s * bi;
            re[i0] = pr * tar - pi * tai;  im[i0] = pr * tai + pi * tar;
            re[i1] = pr * tbr - pi * tbi;  im[i1] = pr * tbi + pi * tbr;
        } else {
            re[0]       *= T_AMP;  im[0]       *= T_AMP;
            re[NCH - 1] *= T_AMP;  im[NCH - 1] *= T_AMP;
        }
        __syncthreads();
    }

    // write C^T in bf16:  crt[n*512 + row] = Re C[row][n]
    for (int j = 0; j < 2; ++j) {
        int n = 2 * t + j;
        crt[(size_t)n * NCH + row] = f2bf(re[n]);
        cit[(size_t)n * NCH + row] = f2bf(im[n]);
    }
}

// ---------------------------------------------------------------------------
// WMMA fragment loaders.
// A (16x32 bf16, row-major src, row stride 512):
//   lane L: M = L&15, klo = (L>>4)*8; elems 0..7 -> K=kk+klo..+7,
//   elems 8..15 -> K=kk+16+klo..+7   (two aligned b128 loads)
// B (32x16 bf16 from transposed [N][K] src):
//   lane L: n = L&15, koff = (L>>4)*16; same K order as A (klo = koff/2)
// ---------------------------------------------------------------------------
__device__ inline FragBF load_a(const unsigned short* rowp, int kk, int klo) {
    FragBF f;
    f.u[0] = *(const uint4*)(rowp + kk + klo);
    f.u[1] = *(const uint4*)(rowp + kk + 16 + klo);
    return f;
}
__device__ inline FragBF load_bt(const unsigned short* colp, int kk, int koff) {
    FragBF f;
    f.u[0] = *(const uint4*)(colp + kk + (koff >> 1));
    f.u[1] = *(const uint4*)(colp + kk + 16 + (koff >> 1));
    return f;
}

#define WMMA_BF16(A, B, C) \
    __builtin_amdgcn_wmma_f32_16x16x32_bf16(false, (A), false, (B), (short)0, (C), false, false)

// ---------------------------------------------------------------------------
// Kernel 3: V = clip((Xr@Cr + XiNeg@Ci) * 1e5, +-5) * 1000, stored bf16.
// 2x2 blocked: one wave owns a 32x32 macro-tile; A frags reused across the
// two N-tiles, B frags across the two M-tiles (2 b128 loads per WMMA).
// ---------------------------------------------------------------------------
__global__ void gemm_mesh_tia_kernel(const unsigned short* __restrict__ xr_bf,
                                     const unsigned short* __restrict__ xin_bf,
                                     const unsigned short* __restrict__ crt,
                                     const unsigned short* __restrict__ cit,
                                     unsigned short* __restrict__ v_bf) {
    const int lane = threadIdx.x & 31;
    const int wave = blockIdx.x * (blockDim.x >> 5) + (threadIdx.x >> 5);
    const int tm2 = wave >> 4;          // 0..127  (32-row macro-tiles in M)
    const int tn2 = wave & 15;          // 0..15   (32-col macro-tiles in N)

    const int m0 = tm2 * 32 + (lane & 15);       // M-tile 0 row for A loads
    const int n0 = tn2 * 32 + (lane & 15);       // N-tile 0 col for B loads
    const int klo  = (lane >> 4) * 8;
    const int koff = (lane >> 4) * 16;

    const unsigned short* ar0 = xr_bf  + (size_t)m0 * NCH;
    const unsigned short* ar1 = ar0 + (size_t)16 * NCH;
    const unsigned short* ai0 = xin_bf + (size_t)m0 * NCH;
    const unsigned short* ai1 = ai0 + (size_t)16 * NCH;
    const unsigned short* br0 = crt + (size_t)n0 * NCH;
    const unsigned short* br1 = br0 + (size_t)16 * NCH;
    const unsigned short* bi0 = cit + (size_t)n0 * NCH;
    const unsigned short* bi1 = bi0 + (size_t)16 * NCH;

    v8f acc00 = {}, acc01 = {}, acc10 = {}, acc11 = {};
    for (int kk = 0; kk < NCH; kk += 32) {
        FragBF aR0 = load_a(ar0, kk, klo), aR1 = load_a(ar1, kk, klo);
        FragBF aI0 = load_a(ai0, kk, klo), aI1 = load_a(ai1, kk, klo);
        FragBF bR0 = load_bt(br0, kk, koff), bR1 = load_bt(br1, kk, koff);
        FragBF bI0 = load_bt(bi0, kk, koff), bI1 = load_bt(bi1, kk, koff);
        acc00 = WMMA_BF16(aR0.v, bR0.v, acc00);
        acc00 = WMMA_BF16(aI0.v, bI0.v, acc00);
        acc01 = WMMA_BF16(aR0.v, bR1.v, acc01);
        acc01 = WMMA_BF16(aI0.v, bI1.v, acc01);
        acc10 = WMMA_BF16(aR1.v, bR0.v, acc10);
        acc10 = WMMA_BF16(aI1.v, bI0.v, acc10);
        acc11 = WMMA_BF16(aR1.v, bR1.v, acc11);
        acc11 = WMMA_BF16(aI1.v, bI1.v, acc11);
    }

    // C/D layout: elem r -> M = tile + r + 8*(lane>>4), N = tile + (lane&15)
    const int nc   = tn2 * 32 + (lane & 15);
    const int mb   = tm2 * 32 + ((lane >> 4) << 3);
    v8f* accs[4] = { &acc00, &acc01, &acc10, &acc11 };
    for (int q = 0; q < 4; ++q) {
        const int mT = mb + (q >> 1) * 16;
        const int nT = nc + (q & 1) * 16;
        for (int r = 0; r < 8; ++r) {
            float y = (*accs[q])[r];
            y = fminf(fmaxf(y * 1e5f, -5.0f), 5.0f) * 1000.0f;  // TIA + Vin
            v_bf[(size_t)(mT + r) * NCH + nT] = f2bf(y);
        }
    }
}

// ---------------------------------------------------------------------------
// Kernel 4: out = V @ G  (f32 accumulate, f32 output), 2x2 blocked.
// ---------------------------------------------------------------------------
__global__ void gemm_crossbar_kernel(const unsigned short* __restrict__ v_bf,
                                     const unsigned short* __restrict__ gt,
                                     float* __restrict__ out) {
    const int lane = threadIdx.x & 31;
    const int wave = blockIdx.x * (blockDim.x >> 5) + (threadIdx.x >> 5);
    const int tm2 = wave >> 4;
    const int tn2 = wave & 15;

    const int m0 = tm2 * 32 + (lane & 15);
    const int n0 = tn2 * 32 + (lane & 15);
    const int klo  = (lane >> 4) * 8;
    const int koff = (lane >> 4) * 16;

    const unsigned short* a0 = v_bf + (size_t)m0 * NCH;
    const unsigned short* a1 = a0 + (size_t)16 * NCH;
    const unsigned short* b0 = gt + (size_t)n0 * NCH;
    const unsigned short* b1 = b0 + (size_t)16 * NCH;

    v8f acc00 = {}, acc01 = {}, acc10 = {}, acc11 = {};
    for (int kk = 0; kk < NCH; kk += 32) {
        FragBF fa0 = load_a(a0, kk, klo), fa1 = load_a(a1, kk, klo);
        FragBF fb0 = load_bt(b0, kk, koff), fb1 = load_bt(b1, kk, koff);
        acc00 = WMMA_BF16(fa0.v, fb0.v, acc00);
        acc01 = WMMA_BF16(fa0.v, fb1.v, acc01);
        acc10 = WMMA_BF16(fa1.v, fb0.v, acc10);
        acc11 = WMMA_BF16(fa1.v, fb1.v, acc11);
    }

    const int nc = tn2 * 32 + (lane & 15);
    const int mb = tm2 * 32 + ((lane >> 4) << 3);
    v8f* accs[4] = { &acc00, &acc01, &acc10, &acc11 };
    for (int q = 0; q < 4; ++q) {
        const int mT = mb + (q >> 1) * 16;
        const int nT = nc + (q & 1) * 16;
        for (int r = 0; r < 8; ++r)
            out[(size_t)(mT + r) * NCH + nT] = (*accs[q])[r];
    }
}

// ---------------------------------------------------------------------------
// Launcher.  Inputs: x_real [4096,512] f32, x_imag [4096,512] f32,
// phases_even [256,256] f32, phases_odd [256,255] f32, states [512,512] f32.
// Output: f32 [4096,512].
// ---------------------------------------------------------------------------
extern "C" void kernel_launch(void* const* d_in, const int* in_sizes, int n_in,
                              void* d_out, int out_size, void* d_ws, size_t ws_size,
                              hipStream_t stream) {
    const float* x_real = (const float*)d_in[0];
    const float* x_imag = (const float*)d_in[1];
    const float* pe     = (const float*)d_in[2];
    const float* po     = (const float*)d_in[3];
    const float* states = (const float*)d_in[4];
    float* out = (float*)d_out;

    char* ws = (char*)d_ws;
    const size_t XB = (size_t)BATCH * NCH * sizeof(unsigned short);  // 4 MB
    const size_t CB = (size_t)NCH * NCH * sizeof(unsigned short);    // 0.5 MB
    unsigned short* xr_bf  = (unsigned short*)(ws);              // [B,512] bf16
    unsigned short* xin_bf = (unsigned short*)(ws + XB);         // [B,512] bf16 (negated imag)
    unsigned short* v_bf   = (unsigned short*)(ws + 2 * XB);     // [B,512] bf16
    unsigned short* crt    = (unsigned short*)(ws + 3 * XB);         // [512,512]^T
    unsigned short* cit    = (unsigned short*)(ws + 3 * XB + CB);
    unsigned short* gt     = (unsigned short*)(ws + 3 * XB + 2 * CB);

    const int nBatchElems = BATCH * NCH;                      // 2,097,152
    convert_x_kernel<<<(nBatchElems + 255) / 256, 256, 0, stream>>>(
        x_real, x_imag, xr_bf, xin_bf, nBatchElems);

    make_gt_kernel<<<(NCH * NCH) / 256, 256, 0, stream>>>(states, gt);

    mesh_build_kernel<<<NCH, 256, 0, stream>>>(pe, po, crt, cit);

    // 2048 macro-tiles of 32x32 over [4096,512]; 8 waves (256 thr) per block
    const int macroTiles = (BATCH / 32) * (NCH / 32);         // 2048
    const int blocks = macroTiles / 8;                        // 256
    gemm_mesh_tia_kernel<<<blocks, 256, 0, stream>>>(xr_bf, xin_bf, crt, cit, v_bf);
    gemm_crossbar_kernel<<<blocks, 256, 0, stream>>>(v_bf, gt, out);
}
